// OPN_60438779789439
// MI455X (gfx1250) — compile-verified, matrix operand order checked by reading, not run on hardware
//
#include <hip/hip_runtime.h>
#include <math.h>

// ---------------------------------------------------------------------------
// Types for CDNA5 WMMA (wave32): D(16x16 f32) = A(16x32 bf16) x B(32x16 bf16) + C
// ---------------------------------------------------------------------------
typedef __attribute__((ext_vector_type(16))) __bf16 v16bf;
typedef __attribute__((ext_vector_type(8)))  float  v8f;

union BF16x16 {
  unsigned short h[16];
  unsigned int   u[8];
  v16bf          v;
};

__device__ __forceinline__ unsigned short f32_to_bf16(float f) {
  unsigned int u = __builtin_bit_cast(unsigned int, f);
  u += 0x7FFFu + ((u >> 16) & 1u);   // round-to-nearest-even
  return (unsigned short)(u >> 16);
}

// A/B fragment K mapping for 16-bit 16x32 operands (ISA 7.12.2):
// VGPR r holds K pair base: r<4 -> 2r, else 16+2(r-4); +8 for lanes 16..31.
__device__ __forceinline__ int fragK(int r, int half) {
  return ((r < 4) ? (r << 1) : (16 + ((r - 4) << 1))) + (half << 3);
}

#define NB 2
#define HH 256
#define WW 256
#define HW (HH * WW)
#define Q4 4096     // 64*64 queries
#define M4 16384    // T*64*64 memory positions

__constant__ float c_mean[3] = {0.485f, 0.456f, 0.406f};
__constant__ float c_std [3] = {0.229f, 0.224f, 0.225f};

// ---------------------------------------------------------------------------
// Preprocess: hole/peel/next_dist, frame composition, normalized 5ch input
// ---------------------------------------------------------------------------
__global__ void k_preproc(const float* __restrict__ frame,
                          const float* __restrict__ valid,
                          const float* __restrict__ dist,
                          float* __restrict__ x5,        // [B,5,H,W]
                          float* __restrict__ peel,      // [B,1,H,W]
                          float* __restrict__ fcomp,     // [B,3,H,W]
                          float* __restrict__ next_dist  // -> d_out tail
                          ) {
  int i = blockIdx.x * blockDim.x + threadIdx.x;
  if (i >= NB * HW) return;
  int b = i / HW, p = i % HW;
  float dv = dist[i];
  float hole = (dv > 0.f) ? 1.f : 0.f;
  float pe   = (dv > 0.f && dv <= 8.f) ? 1.f : 0.f;
  next_dist[i] = fminf(fmaxf(dv - 8.f, 0.f), 9999.f);
  peel[i] = pe;
#pragma unroll
  for (int c = 0; c < 3; ++c) {
    float fr = frame[((size_t)b * 3 + c) * HW + p];
    float fc = (1.f - hole) * fr + hole * c_mean[c];
    fcomp[((size_t)b * 3 + c) * HW + p] = fc;
    x5[((size_t)b * 5 + c) * HW + p] = (fc - c_mean[c]) / c_std[c];
  }
  x5[((size_t)b * 5 + 3) * HW + p] = valid[i];
  x5[((size_t)b * 5 + 4) * HW + p] = hole;
}

// ---------------------------------------------------------------------------
// Bilinear downsample peel 256->64, threshold >= 0.5 -> query mask (f32 0/1)
// ---------------------------------------------------------------------------
__global__ void k_resize_peel(const float* __restrict__ peel,
                              float* __restrict__ qmask) {
  int i = blockIdx.x * blockDim.x + threadIdx.x;
  if (i >= NB * Q4) return;
  int b = i / Q4, p = i % Q4;
  int oy = p / 64, ox = p % 64;
  float sy = (oy + 0.5f) * 4.f - 0.5f;
  float sx = (ox + 0.5f) * 4.f - 0.5f;
  int y0 = (int)floorf(sy), x0 = (int)floorf(sx);
  float wy = sy - y0, wx = sx - x0;
  int y0c = min(max(y0, 0), HH - 1), y1c = min(max(y0 + 1, 0), HH - 1);
  int x0c = min(max(x0, 0), WW - 1), x1c = min(max(x0 + 1, 0), WW - 1);
  const float* pp = peel + (size_t)b * HW;
  float v00 = pp[y0c * WW + x0c], v01 = pp[y0c * WW + x1c];
  float v10 = pp[y1c * WW + x0c], v11 = pp[y1c * WW + x1c];
  float v = v00 * (1 - wy) * (1 - wx) + v01 * (1 - wy) * wx +
            v10 * wy * (1 - wx) + v11 * wy * wx;
  qmask[i] = (v >= 0.5f) ? 1.f : 0.f;
}

// ---------------------------------------------------------------------------
// Memory masks: mm = !mhol (f32), has_mem[b] = any(mm)
// ---------------------------------------------------------------------------
__global__ void k_masks(const unsigned char* __restrict__ mhol,
                        float* __restrict__ mm, float* __restrict__ hasm) {
  __shared__ int any_valid;
  int b = blockIdx.x;
  if (threadIdx.x == 0) any_valid = 0;
  __syncthreads();
  int found = 0;
  for (int i = threadIdx.x; i < M4; i += blockDim.x) {
    unsigned char h = mhol[(size_t)b * M4 + i];
    mm[(size_t)b * M4 + i] = h ? 0.f : 1.f;
    if (!h) found = 1;
  }
  if (found) atomicOr(&any_valid, 1);
  __syncthreads();
  if (threadIdx.x == 0) hasm[b] = any_valid ? 1.f : 0.f;
}

// ---------------------------------------------------------------------------
// Weight conversion: fp32 [Cout, Cin*k*k] -> bf16 [Cout16, K32], zero-padded
// in both rows (to multiple of 16) and K (to multiple of 32).
// ---------------------------------------------------------------------------
__global__ void k_cvt_w(const float* __restrict__ wf, const float* __restrict__ wg,
                        unsigned short* __restrict__ of, unsigned short* __restrict__ og,
                        int Cout, int Cout16, int K, int K32) {
  int i = blockIdx.x * blockDim.x + threadIdx.x;
  if (i >= Cout16 * K32) return;
  int co = i / K32, kk = i % K32;
  bool v = (co < Cout) && (kk < K);
  of[i] = f32_to_bf16(v ? wf[(size_t)co * K + kk] : 0.f);
  og[i] = f32_to_bf16(v ? wg[(size_t)co * K + kk] : 0.f);
}

// ---------------------------------------------------------------------------
// bf16 pre-pack for attention operands:
//   mkT [B][M4][128]  (transposed mkey, pairs along k contiguous for A frags)
//   mvB [B][128][M4]  (mval as bf16, pairs along m contiguous for A frags)
// ---------------------------------------------------------------------------
__global__ void k_prep_attn(const float* __restrict__ mkey,
                            const float* __restrict__ mval,
                            unsigned short* __restrict__ mkT,
                            unsigned short* __restrict__ mvB) {
  int i = blockIdx.x * blockDim.x + threadIdx.x;
  if (i >= NB * 128 * M4) return;
  int m = i % M4; int t = i / M4; int k = t % 128; int b = t / 128;
  mkT[((size_t)b * M4 + m) * 128 + k] = f32_to_bf16(mkey[i]);
  mvB[i] = f32_to_bf16(mval[i]);
}

// ---------------------------------------------------------------------------
// Gated conv, implicit GEMM with bf16 WMMA. One wave computes a
// 16(cout) x 32(spatial) tile: 2 B-fragments share the A (weight) fragments,
// 4 WMMAs per 32-wide K chunk. KSZ and LH=log2(Hout=Wout) are compile-time
// so all index math is shifts/masks/constant magic (no runtime division).
// out = act(Wf*x+bf) * sigmoid(Wg*x+bg)
// ---------------------------------------------------------------------------
template <int KSZ, int LH>
__global__ __launch_bounds__(32)
void k_gconv(const float* __restrict__ in, float* __restrict__ out,
             const unsigned short* __restrict__ Wf, const float* __restrict__ bfv,
             const unsigned short* __restrict__ Wg, const float* __restrict__ bgv,
             int Cin, int Hin, int Win, int Cout,
             int stride, int pad, int dil, int K, int K32, int act) {
  constexpr int KK2 = KSZ * KSZ;
  constexpr int WOUT = 1 << LH;
  constexpr int HWo = 1 << (2 * LH);
  const int mtiles = (NB << (2 * LH)) >> 5;   // 32 spatial columns per wave
  const int mt = blockIdx.x % mtiles;
  const int ct = blockIdx.x / mtiles;
  const int l = threadIdx.x, lo = l & 15, half = l >> 4;

  int bb_[2], oyR[2], oxR[2], oyS[2], oxS[2];
#pragma unroll
  for (int c = 0; c < 2; ++c) {
    int m = mt * 32 + c * 16 + lo;            // NB*HWo is a multiple of 32
    bb_[c] = m >> (2 * LH);
    int rem = m & (HWo - 1);
    oyR[c] = rem >> LH;
    oxR[c] = rem & (WOUT - 1);
    oyS[c] = oyR[c] * stride - pad;
    oxS[c] = oxR[c] * stride - pad;
  }
  const unsigned short* Wfrow = Wf + (size_t)(ct * 16 + lo) * K32;
  const unsigned short* Wgrow = Wg + (size_t)(ct * 16 + lo) * K32;

  v8f accF[2] = {};
  v8f accG[2] = {};

  for (int k0 = 0; k0 < K32; k0 += 32) {
    if (k0 + 32 < K32) {
      __builtin_prefetch((const void*)(Wfrow + k0 + 32), 0, 0);
      __builtin_prefetch((const void*)(Wgrow + k0 + 32), 0, 0);
    }
    BF16x16 af, ag, bcol[2];
#pragma unroll
    for (int r = 0; r < 8; ++r) {
      int kb = k0 + fragK(r, half);
      af.u[r] = *(const unsigned int*)(Wfrow + kb);
      ag.u[r] = *(const unsigned int*)(Wgrow + kb);
#pragma unroll
      for (int e = 0; e < 2; ++e) {
        int kk = kb + e;
        int cin = kk / KK2;                // constant divisor -> magic mul
        int rem = kk - cin * KK2;
        int ky = rem / KSZ;
        int kx = rem - ky * KSZ;
        int dy = ky * dil, dx = kx * dil;
        bool kv = (kk < K);
#pragma unroll
        for (int c = 0; c < 2; ++c) {
          int iy = oyS[c] + dy, ix = oxS[c] + dx;
          float v = 0.f;
          if (kv && iy >= 0 && iy < Hin && ix >= 0 && ix < Win)
            v = in[(((size_t)bb_[c] * Cin + cin) * Hin + iy) * Win + ix];
          bcol[c].h[2 * r + e] = f32_to_bf16(v);
        }
      }
    }
#pragma unroll
    for (int c = 0; c < 2; ++c) {
      accF[c] = __builtin_amdgcn_wmma_f32_16x16x32_bf16(false, af.v, false, bcol[c].v,
                                                        (short)0, accF[c], false, false);
      accG[c] = __builtin_amdgcn_wmma_f32_16x16x32_bf16(false, ag.v, false, bcol[c].v,
                                                        (short)0, accG[c], false, false);
    }
  }

  // epilogue: D element (row=co, col=spatial)
#pragma unroll
  for (int c = 0; c < 2; ++c)
#pragma unroll
    for (int r = 0; r < 8; ++r) {
      int co = ct * 16 + r + 8 * half;
      if (co < Cout) {
        float f = accF[c][r] + bfv[co];
        float g = accG[c][r] + bgv[co];
        if (act) f = (f > 0.f) ? f : 0.2f * f;
        g = 1.f / (1.f + __expf(-g));
        out[(((size_t)bb_[c] * Cout + co) << (2 * LH)) + (oyR[c] << LH) + oxR[c]] = f * g;
      }
    }
}

// ---------------------------------------------------------------------------
// Nearest-neighbor 2x upsample
// ---------------------------------------------------------------------------
__global__ void k_up2(const float* __restrict__ in, float* __restrict__ out,
                      int C, int Hin, int Win) {
  int Hout = Hin * 2, Wout = Win * 2;
  int tot = NB * C * Hout * Wout;
  int i = blockIdx.x * blockDim.x + threadIdx.x;
  if (i >= tot) return;
  int x = i % Wout; int t = i / Wout;
  int y = t % Hout; t /= Hout;           // t = b*C + c
  out[i] = in[((size_t)t * Hin + (y >> 1)) * Win + (x >> 1)];
}

// ---------------------------------------------------------------------------
// Masked attention read, flash-attention style with bf16 WMMA.
// One wave handles 16 queries; streams memory in 32-position chunks.
// Operands pre-packed bf16 so inner loop is pure u32 loads + WMMA.
// ---------------------------------------------------------------------------
__global__ __launch_bounds__(32)
void k_attn(const float* __restrict__ qkey, const float* __restrict__ qval,
            const float* __restrict__ qmask,
            const unsigned short* __restrict__ mkT,   // [B][M4][128] bf16
            const unsigned short* __restrict__ mvB,   // [B][128][M4] bf16
            const float* __restrict__ mm, const float* __restrict__ hasm,
            float* __restrict__ outv) {
  __shared__ float          sS[32][16];
  __shared__ unsigned short sP[32][16];
  __shared__ float          sFac[16];

  const int qtiles = Q4 / 16;
  const int b  = blockIdx.x / qtiles;
  const int qt = blockIdx.x % qtiles;
  const int l = threadIdx.x, lo = l & 15, half = l >> 4;
  const float* mmb = mm + (size_t)b * M4;
  const float inv_sqrt = 0.08838834764831845f; // 1/sqrt(128)

  // Preload qkey B fragments (128 k x 16 q), reused every chunk.
  v16bf qkf[4];
  {
    int q = qt * 16 + lo;
#pragma unroll
    for (int kc = 0; kc < 4; ++kc) {
      BF16x16 t;
#pragma unroll
      for (int r = 0; r < 8; ++r) {
        int kb = kc * 32 + fragK(r, half);
        t.h[2 * r]     = f32_to_bf16(qkey[((size_t)(b * 128 + kb))     * Q4 + q]);
        t.h[2 * r + 1] = f32_to_bf16(qkey[((size_t)(b * 128 + kb + 1)) * Q4 + q]);
      }
      qkf[kc] = t.v;
    }
  }

  v8f accR[8] = {};
  float run_max = -3.0e38f, run_sum = 0.f;

  for (int m0 = 0; m0 < M4; m0 += 32) {
    // --- scores: two 16x16 D tiles over 32 memory rows, K=128 in 4 steps ---
    v8f accS[2] = {};
#pragma unroll
    for (int t = 0; t < 2; ++t) {
      const unsigned short* mkrow = mkT + ((size_t)b * M4 + (m0 + t * 16 + lo)) * 128;
#pragma unroll
      for (int kc = 0; kc < 4; ++kc) {
        BF16x16 af;
#pragma unroll
        for (int r = 0; r < 8; ++r)
          af.u[r] = *(const unsigned int*)(mkrow + kc * 32 + fragK(r, half));
        accS[t] = __builtin_amdgcn_wmma_f32_16x16x32_bf16(false, af.v, false, qkf[kc],
                                                          (short)0, accS[t], false, false);
      }
    }
    // write masked, scaled scores to LDS [32 m][16 q]
#pragma unroll
    for (int t = 0; t < 2; ++t)
#pragma unroll
      for (int r = 0; r < 8; ++r) {
        int rl = t * 16 + r + 8 * half;
        float s = accS[t][r] * inv_sqrt;
        sS[rl][lo] = (mmb[m0 + rl] != 0.f) ? s : -1e9f;
      }
    __syncthreads();

    // --- online softmax per query column (lanes 0..15) ---
    if (l < 16) {
      float cmax = -3.0e38f;
#pragma unroll
      for (int i = 0; i < 32; ++i) cmax = fmaxf(cmax, sS[i][l]);
      float nmax = fmaxf(run_max, cmax);
      float fac = __expf(run_max - nmax);
      float csum = 0.f;
#pragma unroll
      for (int i = 0; i < 32; ++i) {
        float p = __expf(sS[i][l] - nmax);
        csum += p;
        sP[i][l] = f32_to_bf16(p);
      }
      run_sum = run_sum * fac + csum;
      run_max = nmax;
      sFac[l] = fac;
    }
    __syncthreads();

    // --- rescale accumulators and accumulate read = mv @ P ---
    float fac = sFac[lo];
    BF16x16 pf;
#pragma unroll
    for (int r = 0; r < 8; ++r) {
      int kb = fragK(r, half);
      pf.h[2 * r]     = sP[kb][lo];
      pf.h[2 * r + 1] = sP[kb + 1][lo];
    }
#pragma unroll
    for (int d = 0; d < 8; ++d) {
#pragma unroll
      for (int r = 0; r < 8; ++r) accR[d][r] *= fac;
      BF16x16 av;
      const unsigned short* mvp = mvB + ((size_t)(b * 128 + d * 16 + lo)) * M4 + m0;
#pragma unroll
      for (int r = 0; r < 8; ++r)
        av.u[r] = *(const unsigned int*)(mvp + fragK(r, half));
      accR[d] = __builtin_amdgcn_wmma_f32_16x16x32_bf16(false, av.v, false, pf.v,
                                                        (short)0, accR[d], false, false);
    }
    __syncthreads();
  }

  // --- epilogue: out = qval + (acc/run_sum) * qmask * has_mem ---
  if (l < 16) {
    float inv = (run_sum > 0.f) ? 1.f / run_sum : 0.f;
    float gate = qmask[(size_t)b * Q4 + qt * 16 + l] * hasm[b];
    sFac[l] = inv * gate;
  }
  __syncthreads();
  float fac = sFac[lo];
#pragma unroll
  for (int d = 0; d < 8; ++d)
#pragma unroll
    for (int r = 0; r < 8; ++r) {
      int dv = d * 16 + r + 8 * half;
      size_t idx = ((size_t)(b * 128 + dv)) * Q4 + qt * 16 + lo;
      outv[idx] = qval[idx] + accR[d][r] * fac;
    }
}

// ---------------------------------------------------------------------------
// De-normalize decoder output: x = x*STD + MEAN
// ---------------------------------------------------------------------------
__global__ void k_denorm(float* __restrict__ x) {
  int i = blockIdx.x * blockDim.x + threadIdx.x;
  if (i >= NB * 3 * HW) return;
  int c = (i / HW) % 3;
  x[i] = x[i] * c_std[c] + c_mean[c];
}

// ---------------------------------------------------------------------------
// Final composition: comp = clip((1-peel)*frame_comp + peel*pred, 0, 1)
// ---------------------------------------------------------------------------
__global__ void k_final(const float* __restrict__ pred,
                        const float* __restrict__ fcomp,
                        const float* __restrict__ peel,
                        float* __restrict__ comp) {
  int i = blockIdx.x * blockDim.x + threadIdx.x;
  if (i >= NB * 3 * HW) return;
  int p = i % HW;
  int b = i / (3 * HW);
  float pe = peel[(size_t)b * HW + p];
  float v = (1.f - pe) * fcomp[i] + pe * pred[i];
  comp[i] = fminf(fmaxf(v, 0.f), 1.f);
}

// ---------------------------------------------------------------------------
// Host launcher
// ---------------------------------------------------------------------------
extern "C" void kernel_launch(void* const* d_in, const int* in_sizes, int n_in,
                              void* d_out, int out_size, void* d_ws, size_t ws_size,
                              hipStream_t stream) {
  (void)in_sizes; (void)n_in; (void)out_size; (void)ws_size;
  const float* mkey = (const float*)d_in[0];
  const float* mval = (const float*)d_in[1];
  const unsigned char* mhol = (const unsigned char*)d_in[2];
  const float* frame = (const float*)d_in[3];
  const float* valid = (const float*)d_in[4];
  const float* dist  = (const float*)d_in[5];
  // params pytree flattened with sorted dict keys:
  // outer: d2,d21,d32,d3a,d3b,d3c,d3d,e12,e2,e23,e3a,e3b,e3c,e3d,key3,val3
  // inner: bf,bg,wf,wg
  auto P = [&](int pidx, int which) -> const float* {
    return (const float*)d_in[6 + pidx * 4 + which];
  };

  char* ws = (char*)d_ws;
  float* bufA  = (float*)(ws + 0);            // 33.6 MB ping
  float* bufB  = (float*)(ws + 33554432);     // 33.6 MB pong
  float* QK    = (float*)(ws + 67108864);     // 4.2 MB
  float* QV    = (float*)(ws + 71303168);     // 4.2 MB
  float* RD    = (float*)(ws + 75497472);     // 4.2 MB
  float* PEEL  = (float*)(ws + 79691776);
  float* FCOMP = (float*)(ws + 80216064);
  float* QMASK = (float*)(ws + 81788928);
  float* MM    = (float*)(ws + 81821696);
  float* HASM  = (float*)(ws + 81952768);
  unsigned short* WBASE = (unsigned short*)(ws + 81953024);  // ~7 MB bf16 weights
  unsigned short* MKT   = (unsigned short*)(ws + 89128960);  // 8.4 MB
  unsigned short* MVB   = (unsigned short*)(ws + 97517568);  // 8.4 MB

  float* out_comp = (float*)d_out;
  float* out_nd   = (float*)d_out + NB * 3 * HW;

  // Execution-order layer metadata (pidx = sorted param index)
  const int PIDX[16] = {7, 8, 9,10,11,12,13,14,15, 6, 5, 4, 3, 2, 0, 1};
  const int CIN [16] = {5,64,64,128,128,128,128,128,128,128,128,128,128,128,64,64};
  const int COUT[16] = {64,64,128,128,128,128,128,128,128,128,128,128,128,64,64,3};
  const int KS  [16] = {5,3,3,3,3,3,3,3,3,3,3,3,3,3,3,5};
  const int ST  [16] = {2,1,2,1,1,1,1,1,1,1,1,1,1,1,1,1};
  const int PD  [16] = {2,1,1,1,2,4,8,1,1,8,4,2,1,1,1,2};
  const int DL  [16] = {1,1,1,1,2,4,8,1,1,8,4,2,1,1,1,1};
  const int HIN [16] = {256,128,128,64,64,64,64,64,64,64,64,64,64,128,128,256};
  const int HOUT[16] = {128,128,64,64,64,64,64,64,64,64,64,64,64,128,128,256};
  const int LH  [16] = {7,7,6,6,6,6,6,6,6,6,6,6,6,7,7,8};
  const int ACT [16] = {1,1,1,1,1,1,1,0,0,1,1,1,1,1,1,0};

  // Convert all weights to padded bf16 matrices (rows padded to 16).
  unsigned short* Wf[16]; unsigned short* Wg[16];
  int Kk[16], K32[16];
  {
    unsigned short* wp = WBASE;
    for (int i = 0; i < 16; ++i) {
      int K = CIN[i] * KS[i] * KS[i];
      int k32 = ((K + 31) / 32) * 32;
      int c16 = ((COUT[i] + 15) / 16) * 16;
      Kk[i] = K; K32[i] = k32;
      Wf[i] = wp; wp += (size_t)c16 * k32;
      Wg[i] = wp; wp += (size_t)c16 * k32;
      int tot = c16 * k32;
      k_cvt_w<<<dim3((tot + 255) / 256), dim3(256), 0, stream>>>(
          P(PIDX[i], 2), P(PIDX[i], 3), Wf[i], Wg[i], COUT[i], c16, K, k32);
    }
  }

  // Preprocess + masks + attention operand pre-pack
  k_preproc<<<dim3((NB * HW + 255) / 256), dim3(256), 0, stream>>>(
      frame, valid, dist, bufA, PEEL, FCOMP, out_nd);
  k_resize_peel<<<dim3((NB * Q4 + 255) / 256), dim3(256), 0, stream>>>(PEEL, QMASK);
  k_masks<<<dim3(NB), dim3(256), 0, stream>>>(mhol, MM, HASM);
  k_prep_attn<<<dim3((NB * 128 * M4 + 255) / 256), dim3(256), 0, stream>>>(
      mkey, mval, MKT, MVB);

  auto gconv = [&](int i, const float* in, float* out) {
    int mtiles = (NB * HOUT[i] * HOUT[i]) / 32;
    int ctiles = (COUT[i] + 15) / 16;
    dim3 g(mtiles * ctiles), t(32);
#define GARGS in, out, Wf[i], P(PIDX[i], 0), Wg[i], P(PIDX[i], 1), \
              CIN[i], HIN[i], HIN[i], COUT[i], ST[i], PD[i], DL[i], Kk[i], K32[i], ACT[i]
    if (KS[i] == 5 && LH[i] == 7)      k_gconv<5, 7><<<g, t, 0, stream>>>(GARGS);
    else if (KS[i] == 3 && LH[i] == 7) k_gconv<3, 7><<<g, t, 0, stream>>>(GARGS);
    else if (KS[i] == 3 && LH[i] == 6) k_gconv<3, 6><<<g, t, 0, stream>>>(GARGS);
    else                               k_gconv<5, 8><<<g, t, 0, stream>>>(GARGS);
#undef GARGS
  };

  // Encoder
  gconv(0, bufA, bufB);   // e12  -> [B,64,128,128]
  gconv(1, bufB, bufA);   // e2
  gconv(2, bufA, bufB);   // e23  -> [B,128,64,64]
  gconv(3, bufB, bufA);   // e3a
  gconv(4, bufA, bufB);   // e3b
  gconv(5, bufB, bufA);   // e3c
  gconv(6, bufA, bufB);   // e3d
  gconv(7, bufB, QK);     // key3
  gconv(8, bufB, QV);     // val3

  // Attention read
  k_attn<<<dim3(NB * (Q4 / 16)), dim3(32), 0, stream>>>(
      QK, QV, QMASK, MKT, MVB, MM, HASM, RD);

  // Decoder
  gconv(9,  RD,   bufA);  // d3d
  gconv(10, bufA, bufB);  // d3c
  gconv(11, bufB, bufA);  // d3b
  gconv(12, bufA, bufB);  // d3a
  k_up2<<<dim3((NB * 128 * 128 * 128 + 255) / 256), dim3(256), 0, stream>>>(
      bufB, bufA, 128, 64, 64);                     // -> [B,128,128,128]
  gconv(13, bufA, bufB);  // d32 -> [B,64,128,128]
  gconv(14, bufB, bufA);  // d2
  k_up2<<<dim3((NB * 64 * 256 * 256 + 255) / 256), dim3(256), 0, stream>>>(
      bufA, bufB, 64, 128, 128);                    // -> [B,64,256,256]
  gconv(15, bufB, bufA);  // d21 -> [B,3,256,256] (pred, normalized space)

  k_denorm<<<dim3((NB * 3 * HW + 255) / 256), dim3(256), 0, stream>>>(bufA);

  // Final composition
  k_final<<<dim3((NB * 3 * HW + 255) / 256), dim3(256), 0, stream>>>(
      bufA, FCOMP, PEEL, out_comp);
}
